// fVSRN_48842368090678
// MI455X (gfx1250) — matrix operand order repaired and checked by reading, not compiled, not run on hardware
//
#include <hip/hip_runtime.h>

#define LFREQ   6
#define GRIDSZ  64
#define NFEAT   16
#define NODES   128
#define DIN_PAD 64
#define TILE    256
#define BLOCK   256

typedef _Float16 h4   __attribute__((ext_vector_type(4)));
typedef _Float16 h8   __attribute__((ext_vector_type(8)));
typedef _Float16 h16  __attribute__((ext_vector_type(16)));
typedef float    f4v  __attribute__((ext_vector_type(4)));
typedef float    f8v  __attribute__((ext_vector_type(8)));

// ---------------------------------------------------------------------------
// Pre-pass: [C, D, H, W] -> [D, H, W, C] so each trilinear tap is one 64B read
// ---------------------------------------------------------------------------
__global__ void fVSRN_transpose_grid(const float* __restrict__ g,
                                     float* __restrict__ t) {
    int idx = blockIdx.x * blockDim.x + threadIdx.x;       // voxel id
    const int NVOX = GRIDSZ * GRIDSZ * GRIDSZ;
    if (idx >= NVOX) return;
    #pragma unroll
    for (int c = 0; c < NFEAT; ++c)
        t[idx * NFEAT + c] = g[c * NVOX + idx];
}

__device__ __forceinline__ h16 cat16(h8 lo, h8 hi) {
    return __builtin_shufflevector(lo, hi, 0, 1, 2, 3, 4, 5, 6, 7,
                                   8, 9, 10, 11, 12, 13, 14, 15);
}

// ---------------------------------------------------------------------------
// One fused GEMM layer with snake activation, IN PLACE in LDS.
// Safe because each wave reads A-fragments only from its own 32 rows (hoisted
// to VGPRs at M-tile start) and writes only those rows; same-wave LDS ops are
// in order, and a workgroup barrier separates layers.
//   - first WMMA of each chain takes literal-0 C (no init movs); bias added
//     in the epilogue
//   - per K-chunk: load ALL 4 B-fragments, then 4 independent WMMAs
// Fragment layouts per CDNA5 ISA 7.12.2 (wave32):
//   A (16x32 f16): lane l16*16+ln holds M=ln; halves 0..7 = K kb+l16*8..+7,
//                  halves 8..15 = K kb+16+l16*8..+7   -> two b128 LDS loads
//   B (32x16 f16): lane holds N=ln; halves 0..7 = K kb+l16*8..+7,
//                  halves 8..15 = K kb+16+l16*8..+7   -> two b128 LDS loads
//   C (16x16 f32): VGPR r -> row r + 8*l16, col ln
// ---------------------------------------------------------------------------
template <int NK>
__device__ __forceinline__ void gemm_snake(_Float16* act,
                                           const _Float16* __restrict__ w,
                                           const float* __restrict__ sBias) {
    const int lane = threadIdx.x & 31;
    const int wave = threadIdx.x >> 5;
    const int l16  = lane >> 4;
    const int ln   = lane & 15;

    const f8v zeroC = {0.f, 0.f, 0.f, 0.f, 0.f, 0.f, 0.f, 0.f};

    #pragma unroll
    for (int mi = 0; mi < 2; ++mi) {                  // 16 M-tiles / 8 waves
        const int mt = wave * 2 + mi;
        const _Float16* ap = act + (mt * 16 + ln) * NODES + l16 * 8;

        h16 Af[NK];
        #pragma unroll
        for (int kc = 0; kc < NK; ++kc) {
            h8 a0 = *(const h8*)(ap + kc * 32);
            h8 a1 = *(const h8*)(ap + kc * 32 + 16);
            Af[kc] = cat16(a0, a1);
        }

        #pragma unroll
        for (int ng = 0; ng < 2; ++ng) {              // 2 groups of 4 N-tiles
            f8v acc[4];

            #pragma unroll
            for (int kc = 0; kc < NK; ++kc) {
                h16 Bf[4];
                #pragma unroll
                for (int j = 0; j < 4; ++j) {         // batch the 8 ds_load_b128
                    const _Float16* bp =
                        w + ((ng * 4 + j) * 16 + ln) * NODES + kc * 32 + l16 * 8;
                    h8 b0 = *(const h8*)(bp);
                    h8 b1 = *(const h8*)(bp + 16);
                    Bf[j] = cat16(b0, b1);
                }
                #pragma unroll
                for (int j = 0; j < 4; ++j) {         // 4 independent chains
                    acc[j] = __builtin_amdgcn_wmma_f32_16x16x32_f16(
                        false, Af[kc], false, Bf[j], (short)0,
                        (kc == 0) ? zeroC : acc[j], false, false);
                }
            }

            #pragma unroll
            for (int j = 0; j < 4; ++j) {
                const float bv = sBias[(ng * 4 + j) * 16 + ln];
                #pragma unroll
                for (int r = 0; r < 8; ++r) {
                    float v = acc[j][r] + bv;
                    float s = __sinf(v);
                    float o = 0.5f * v + s * s;       // snake
                    act[(mt * 16 + l16 * 8 + r) * NODES + (ng * 4 + j) * 16 + ln] =
                        (_Float16)o;
                }
            }
        }
    }
}

// ---------------------------------------------------------------------------
// Fused neural-field kernel: PE + trilinear grid sample + 4 snake layers + out
// ---------------------------------------------------------------------------
__global__ __launch_bounds__(BLOCK, 4)
void fVSRN_fused(const float* __restrict__ x,
                 const float* __restrict__ gridT,
                 const float* __restrict__ W0,
                 const float* __restrict__ b0,
                 const float* __restrict__ Wh,
                 const float* __restrict__ bh,
                 const float* __restrict__ Wout,
                 const float* __restrict__ bout,
                 float* __restrict__ out,
                 int npts) {
    __shared__ __align__(16) _Float16 sW[NODES * NODES];   // 32 KB weights
    __shared__ __align__(16) _Float16 sAct[TILE * NODES];  // 64 KB activations
    __shared__ float sBias[NODES];                         // per-layer bias
    __shared__ float sWout[NODES];                         // output weights

    const int tid = threadIdx.x;
    const int p0  = blockIdx.x * TILE;
    const int p   = p0 + tid;

    // ---- stage W0 [128 x 52] -> sW (stride NODES, K padded to 64)
    {
        const int row = tid >> 1;
        const int c0  = (tid & 1) * 64;
        const float* wsrc = W0 + row * (LFREQ * 2 * 3 + NFEAT);    // 52 cols
        _Float16* wdst = sW + row * NODES + c0;
        if (c0 == 0) {
            #pragma unroll
            for (int k = 0; k < 52; ++k) wdst[k] = (_Float16)wsrc[k];
            #pragma unroll
            for (int k = 52; k < 64; ++k) wdst[k] = (_Float16)0.f;
        }
        if (tid < NODES) { sBias[tid] = b0[tid]; sWout[tid] = Wout[tid]; }
    }

    // ---- phase 1: positional encoding + trilinear feature gather -> sAct
    {
        float cx = 0.f, cy = 0.f, cz = 0.f;
        if (p < npts) { cx = x[3 * p]; cy = x[3 * p + 1]; cz = x[3 * p + 2]; }
        _Float16* row = sAct + tid * NODES;

        float f = 3.14159265358979323846f;
        #pragma unroll
        for (int l = 0; l < LFREQ; ++l) {
            row[l * 6 + 0] = (_Float16)__sinf(f * cx);
            row[l * 6 + 1] = (_Float16)__sinf(f * cy);
            row[l * 6 + 2] = (_Float16)__sinf(f * cz);
            row[l * 6 + 3] = (_Float16)__cosf(f * cx);
            row[l * 6 + 4] = (_Float16)__cosf(f * cy);
            row[l * 6 + 5] = (_Float16)__cosf(f * cz);
            f *= 2.f;
        }

        // trilinear sample, align_corners=True
        float fx = fminf(fmaxf((cx + 1.f) * 0.5f * (GRIDSZ - 1), 0.f), (float)(GRIDSZ - 1));
        float fy = fminf(fmaxf((cy + 1.f) * 0.5f * (GRIDSZ - 1), 0.f), (float)(GRIDSZ - 1));
        float fz = fminf(fmaxf((cz + 1.f) * 0.5f * (GRIDSZ - 1), 0.f), (float)(GRIDSZ - 1));
        int x0 = min((int)fx, GRIDSZ - 1), x1 = min(x0 + 1, GRIDSZ - 1);
        int y0 = min((int)fy, GRIDSZ - 1), y1 = min(y0 + 1, GRIDSZ - 1);
        int z0 = min((int)fz, GRIDSZ - 1), z1 = min(z0 + 1, GRIDSZ - 1);
        float wx1 = fx - (float)x0, wx0 = 1.f - wx1;
        float wy1 = fy - (float)y0, wy0 = 1.f - wy1;
        float wz1 = fz - (float)z0, wz0 = 1.f - wz1;

        const int xs[2] = {x0, x1}, ys[2] = {y0, y1}, zs[2] = {z0, z1};
        const float wxs[2] = {wx0, wx1}, wys[2] = {wy0, wy1}, wzs[2] = {wz0, wz1};

        float acc[NFEAT];
        #pragma unroll
        for (int c = 0; c < NFEAT; ++c) acc[c] = 0.f;

        #pragma unroll
        for (int dz = 0; dz < 2; ++dz)
            #pragma unroll
            for (int dy = 0; dy < 2; ++dy)
                #pragma unroll
                for (int dx = 0; dx < 2; ++dx) {
                    float wgt = wzs[dz] * wys[dy] * wxs[dx];
                    const float* g = gridT +
                        (size_t)((zs[dz] * GRIDSZ + ys[dy]) * GRIDSZ + xs[dx]) * NFEAT;
                    #pragma unroll
                    for (int c = 0; c < NFEAT; ++c) acc[c] += wgt * g[c];
                }

        #pragma unroll
        for (int c = 0; c < NFEAT; ++c) row[36 + c] = (_Float16)acc[c];
        #pragma unroll
        for (int k = 52; k < DIN_PAD; ++k) row[k] = (_Float16)0.f;
    }
    __syncthreads();

    // ---- layer 0: [TILE x 64] x [64 x 128] in-place (2 K-chunks)
    gemm_snake<2>(sAct, sW, sBias);
    __syncthreads();

    // ---- 3 hidden layers, in place, reload weights each layer
    //      vectorized staging: float4 global loads -> packed f16 -> b64 stores
    #pragma unroll
    for (int i = 0; i < 3; ++i) {
        const int row = tid >> 1;
        const int c0  = (tid & 1) * 64;
        const f4v* ws4 = (const f4v*)(Wh + (size_t)i * NODES * NODES +
                                      row * NODES + c0);           // 256B aligned
        h4* wd4 = (h4*)(sW + row * NODES + c0);
        #pragma unroll
        for (int k = 0; k < 16; ++k) {
            f4v v = ws4[k];
            h4 h;
            h[0] = (_Float16)v[0]; h[1] = (_Float16)v[1];
            h[2] = (_Float16)v[2]; h[3] = (_Float16)v[3];
            wd4[k] = h;
        }
        if (tid < NODES) sBias[tid] = bh[i * NODES + tid];
        __syncthreads();

        gemm_snake<4>(sAct, sW, sBias);
        __syncthreads();
    }

    // ---- output layer: 128 -> 1 dot product per point
    {
        float s = bout[0];
        const _Float16* row = sAct + tid * NODES;
        #pragma unroll
        for (int k = 0; k < NODES; ++k) s += (float)row[k] * sWout[k];
        if (p < npts) out[p] = s;
    }
}

// ---------------------------------------------------------------------------
extern "C" void kernel_launch(void* const* d_in, const int* in_sizes, int n_in,
                              void* d_out, int out_size, void* d_ws, size_t ws_size,
                              hipStream_t stream) {
    const float* x    = (const float*)d_in[0];
    const float* grid = (const float*)d_in[1];
    const float* W0   = (const float*)d_in[2];
    const float* b0   = (const float*)d_in[3];
    const float* Wh   = (const float*)d_in[4];
    const float* bh   = (const float*)d_in[5];
    const float* Wout = (const float*)d_in[6];
    const float* bout = (const float*)d_in[7];
    float* out   = (float*)d_out;
    float* gridT = (float*)d_ws;                    // 16 MB channel-last grid

    const int npts = in_sizes[0] / 3;
    const int nvox = GRIDSZ * GRIDSZ * GRIDSZ;

    fVSRN_transpose_grid<<<(nvox + 255) / 256, 256, 0, stream>>>(grid, gridT);

    const int nblk = (npts + TILE - 1) / TILE;
    fVSRN_fused<<<nblk, BLOCK, 0, stream>>>(x, gridT, W0, b0, Wh, bh, Wout, bout,
                                            out, npts);
}
